// DemodulatedConvolution_18897856102547
// MI455X (gfx1250) — matrix operand order repaired and checked by reading, not compiled, not run on hardware
//
#include <hip/hip_runtime.h>

// Problem constants
#define B_   16
#define H_   256
#define W_   256
#define CIN  64
#define FIL  64
#define LAT  512

// LDS layout (bytes): input hi/lo (2*25344 bf16) + weight hi/lo (2*36864 bf16) + style (64 f32)
#define IN_ELEMS  (3*132*64)   // 25344 bf16 per half
#define WT_ELEMS  (9*64*64)    // 36864 bf16 per half
#define LDS_BYTES (2*IN_ELEMS*2 + 2*WT_ELEMS*2 + 64*4)   // 249088 B  (< 320KB WGP LDS)

typedef __attribute__((ext_vector_type(16))) __bf16        v16bf;
typedef __attribute__((ext_vector_type(8)))  float         v8f;
typedef __attribute__((ext_vector_type(4)))  unsigned int  u32x4;

union Frag  { v16bf bf; u32x4 u[2]; };
union Pack4 { __bf16 b[4]; uint2 u; };

__device__ __forceinline__ void split_bf16(float x, __bf16& hi, __bf16& lo) {
  hi = (__bf16)x;                 // RNE truncation to bf16
  lo = (__bf16)(x - (float)hi);   // residual
}

// ---------------------------------------------------------------------------
// Prep: style[n][c] = latent[n]·dense_w[:,c]/sqrt(512) + dense_b[c]
//       demod[n][f] = rsqrt( sum_ci style^2 * sum_tap w_scaled^2 + 1e-8 )
// Tiny (≈2 MFLOP total): one block per batch, 64 threads, exact fp32.
// ---------------------------------------------------------------------------
__global__ void prep_kernel(const float* __restrict__ latent,
                            const float* __restrict__ dw,
                            const float* __restrict__ db,
                            const float* __restrict__ ck,
                            float* __restrict__ style_ws,
                            float* __restrict__ demod_ws) {
  const float RS = 0.04419417382415922f;   // 1/sqrt(512)
  const float WS = 0.05892556509887896f;   // sqrt(2/(3*3*64))
  const int n = blockIdx.x, t = threadIdx.x;
  __shared__ float st[64];
  float acc = 0.f;
  for (int k = 0; k < LAT; ++k)
    acc = fmaf(latent[n*LAT + k], dw[k*CIN + t], acc);
  const float s = acc * RS + db[t];
  st[t] = s;
  style_ws[n*64 + t] = s;
  __syncthreads();
  float d = 0.f;
  for (int ci = 0; ci < CIN; ++ci) {
    const float s2 = st[ci] * st[ci];
    float kq = 0.f;
    for (int tap = 0; tap < 9; ++tap) {
      const float w = ck[tap*4096 + ci*64 + t] * WS;
      kq = fmaf(w, w, kq);
    }
    d = fmaf(s2, kq, d);
  }
  demod_ws[n*64 + t] = rsqrtf(d + 1e-8f);
}

// ---------------------------------------------------------------------------
// Main conv: implicit GEMM, M = B*H*W, N = 64, K = 9*64 = 576.
// Block = (n, h, 128-wide W segment) -> 128 px * 64 f.  8 wave32s, each wave
// computes a 16px x 64f tile with v_wmma_f32_16x16x32_bf16 in hi/lo split
// (3 products per bf16 WMMA step => ~fp32 accuracy at bf16 matrix rate).
// ---------------------------------------------------------------------------
__global__ void __launch_bounds__(256)
conv_kernel(const float* __restrict__ fm,
            const float* __restrict__ ck,
            const float* __restrict__ bias,
            const float* __restrict__ style_ws,
            const float* __restrict__ demod_ws,
            float* __restrict__ out) {
  extern __shared__ __align__(16) char smem[];
  __bf16* inHi = (__bf16*)smem;                          // [3][132][64]
  __bf16* inLo = inHi + IN_ELEMS;
  __bf16* wHi  = (__bf16*)(smem + 2*IN_ELEMS*2);         // WMMA-B-fragment order
  __bf16* wLo  = wHi + WT_ELEMS;
  float*  sstyle = (float*)(smem + 2*IN_ELEMS*2 + 2*WT_ELEMS*2);

  const int tid = threadIdx.x;
  const int bid = blockIdx.x;
  const int n   = bid >> 9;            // 512 blocks per batch image
  const int h   = (bid & 511) >> 1;
  const int w0  = (bid & 1) * 128;
  const float WS = 0.05892556509887896f;

  if (tid < 64) sstyle[tid] = style_ws[n*64 + tid];
  __syncthreads();

  // --- Stage scaled weights into LDS, pre-split & pre-swizzled into the
  //     exact bf16 B-matrix fragment layout (per tap, K-tile, N-tile):
  //     lane = ((kk>>3)&1 ? 16+f%16 : f%16), within-lane element = (kk&7) + 8*(kk>>4).
  for (int e = tid; e < WT_ELEMS; e += 256) {
    const int tap = e >> 12;
    const int ci  = (e >> 6) & 63;
    const int f   = e & 63;
    const float v = ck[e] * WS;        // src layout [kh][kw][ci][f] == flat e
    __bf16 hi, lo; split_bf16(v, hi, lo);
    const int kt   = ci >> 5;
    const int kk   = ci & 31;
    const int lane = (((kk >> 3) & 1) ? 16 + (f & 15) : (f & 15));
    const int dst  = (((tap*2 + kt)*4 + (f >> 4))*32 + lane)*16 + ((kk & 7) + ((kk >> 4) << 3));
    wHi[dst] = hi; wLo[dst] = lo;
  }

  // --- Stage modulated input halo tile (rows h-1..h+1, cols w0-1..w0+128),
  //     fp32 * style -> split to bf16 hi/lo.  SAME padding => zeros OOB.
  for (int i = tid; i < 3*130*16; i += 256) {
    const int r   = i / 2080;          // 130*16
    const int rem = i - r*2080;
    const int c   = rem >> 4;          // 0..129
    const int ci  = (rem & 15) << 2;
    const int gh  = h - 1 + r;
    const int gw  = w0 - 1 + c;
    float4 x = make_float4(0.f, 0.f, 0.f, 0.f);
    if (gh >= 0 && gh < H_ && gw >= 0 && gw < W_)
      x = *(const float4*)(fm + ((((long)n*H_ + gh)*W_ + gw)*CIN + ci));
    const float xs[4] = {x.x, x.y, x.z, x.w};
    Pack4 ph, pl;
    #pragma unroll
    for (int k = 0; k < 4; ++k)
      split_bf16(xs[k] * sstyle[ci + k], ph.b[k], pl.b[k]);
    const int base = (r*132 + c)*64 + ci;
    *(uint2*)(inHi + base) = ph.u;
    *(uint2*)(inLo + base) = pl.u;
  }
  __syncthreads();

  const int lane = tid & 31;
  const int m16  = lane & 15;
  const int half = lane >> 4;
  const int pixBase = (tid >> 5) * 16;   // wave's 16-pixel group

  v8f acc[4] = {};

  for (int tap = 0; tap < 9; ++tap) {
    const int kh = tap / 3, kw = tap - kh*3;
    // A-fragment row: 16-bit A layout => lanes 0-15 hold K=0..7 & 16..23,
    // lanes 16-31 hold K=8..15 & 24..31 (channel-contiguous b128 runs).
    const int aRow = (kh*132 + pixBase + m16 + kw)*64;
    #pragma unroll
    for (int kt = 0; kt < 2; ++kt) {
      const int ci0 = kt*32 + 8*half;
      Frag aH, aL;
      aH.u[0] = *(const u32x4*)(inHi + aRow + ci0);
      aH.u[1] = *(const u32x4*)(inHi + aRow + ci0 + 16);
      aL.u[0] = *(const u32x4*)(inLo + aRow + ci0);
      aL.u[1] = *(const u32x4*)(inLo + aRow + ci0 + 16);
      #pragma unroll
      for (int nt = 0; nt < 4; ++nt) {
        const int fb = ((tap*2 + kt)*4 + nt)*512 + lane*16;
        Frag bH, bL;
        bH.u[0] = *(const u32x4*)(wHi + fb);
        bH.u[1] = *(const u32x4*)(wHi + fb + 8);
        bL.u[0] = *(const u32x4*)(wLo + fb);
        bL.u[1] = *(const u32x4*)(wLo + fb + 8);
        // hi*hi + hi*lo + lo*hi  (lo*lo ~2^-16, dropped)
        acc[nt] = __builtin_amdgcn_wmma_f32_16x16x32_bf16(false, aH.bf, false, bH.bf, (short)0, acc[nt], false, false);
        acc[nt] = __builtin_amdgcn_wmma_f32_16x16x32_bf16(false, aH.bf, false, bL.bf, (short)0, acc[nt], false, false);
        acc[nt] = __builtin_amdgcn_wmma_f32_16x16x32_bf16(false, aL.bf, false, bH.bf, (short)0, acc[nt], false, false);
      }
    }
  }

  // --- Epilogue: demodulate + bias, stage in LDS (reuse input region) for
  //     coalesced b128 global stores.
  __syncthreads();
  float* stg = (float*)smem;   // 128px * 64f * 4B = 32KB, fits in input region
  #pragma unroll
  for (int nt = 0; nt < 4; ++nt) {
    const int f = nt*16 + m16;
    const float dm = demod_ws[n*64 + f];
    const float bs = bias[f];
    #pragma unroll
    for (int r = 0; r < 8; ++r) {
      const int pix = pixBase + r + 8*half;   // C/D layout: M = r + 8*(lane/16)
      stg[pix*64 + f] = acc[nt][r] * dm + bs;
    }
  }
  __syncthreads();
  const long outBase = (((long)n*H_ + h)*W_ + w0)*CIN;
  float4* ov = (float4*)(out + outBase);
  const float4* sv = (const float4*)stg;
  for (int i = tid; i < 2048; i += 256) ov[i] = sv[i];
}

// ---------------------------------------------------------------------------
extern "C" void kernel_launch(void* const* d_in, const int* in_sizes, int n_in,
                              void* d_out, int out_size, void* d_ws, size_t ws_size,
                              hipStream_t stream) {
  const float* fm     = (const float*)d_in[0];  // feature_map
  const float* latent = (const float*)d_in[1];
  const float* dw     = (const float*)d_in[2];  // dense_w
  const float* db     = (const float*)d_in[3];  // dense_b
  const float* ck     = (const float*)d_in[4];  // conv_kernel
  const float* bias   = (const float*)d_in[5];
  float* out = (float*)d_out;

  float* style_ws = (float*)d_ws;          // 16*64 floats
  float* demod_ws = style_ws + B_*FIL;     // 16*64 floats

  prep_kernel<<<B_, 64, 0, stream>>>(latent, dw, db, ck, style_ws, demod_ws);
  conv_kernel<<<B_*H_*(W_/128), 256, LDS_BYTES, stream>>>(fm, ck, bias,
                                                          style_ws, demod_ws, out);
}
